// Listener_77988016161119
// MI455X (gfx1250) — matrix operand order repaired and checked by reading, not compiled, not run on hardware
//
#include <hip/hip_runtime.h>
#include <hip/hip_bf16.h>

typedef __attribute__((ext_vector_type(16))) __bf16 v16bf;
typedef __attribute__((ext_vector_type(8)))  float  v8f;

union FragBF { v16bf v; int4 i[2]; };

__device__ __forceinline__ unsigned short f2bf(float x) {
    union { float f; unsigned int u; } v; v.f = x;
    unsigned int u = v.u;
    unsigned int r = u + 0x7FFFu + ((u >> 16) & 1u);   // round-to-nearest-even
    return (unsigned short)(r >> 16);
}
__device__ __forceinline__ float bf2f(unsigned short h) {
    union { unsigned int u; float f; } v; v.u = ((unsigned int)h) << 16;
    return v.f;
}
__device__ __forceinline__ float sigm(float x) {
    return 1.f / (1.f + __expf(-x));
}

// ---------------------------------------------------------------------------
// x0 = (inputs @ dem_w + dem_b) * SCI * scale + PET(t,d)   -> bf16 [b*T+t][512]
// ---------------------------------------------------------------------------
__global__ void embed_kernel(const float* __restrict__ inp,
                             const float* __restrict__ dem_w,
                             const float* __restrict__ dem_b,
                             const float* __restrict__ scale,
                             unsigned short* __restrict__ X0, int T)
{
    int idx = blockIdx.x * blockDim.x + threadIdx.x;   // row*512 + d
    if (idx >= 32 * T * 512) return;
    int d   = idx & 511;
    int row = idx >> 9;                                // b*T + t
    int t   = row % T;
    const float* ip = inp + (size_t)row * 40;
    const float* wp = dem_w + d;
    float acc = dem_b[d];
#pragma unroll 8
    for (int k = 0; k < 40; k++) acc += ip[k] * wp[(size_t)k * 512];
    float sci  = 71.55417527999327f * scale[0];        // sqrt(512/40)*20
    float freq = __expf(-(float)((d >> 1) << 1) * (9.210340371976184f / 512.f));
    float ang  = (float)t * freq;
    float pet  = (d & 1) ? __cosf(ang) : __sinf(ang);
    X0[idx] = f2bf(acc * sci + pet);
}

// ---------------------------------------------------------------------------
// Pack fp32 W[K][N] -> bf16 Wt[n][K] (K-contiguous for WMMA B fragments)
// ---------------------------------------------------------------------------
__global__ void packT_kernel(const float* __restrict__ W,
                             unsigned short* __restrict__ Wt, int K, int N)
{
    int idx = blockIdx.x * blockDim.x + threadIdx.x;
    if (idx >= K * N) return;
    int k = idx % K, n = idx / K;
    Wt[idx] = f2bf(W[(size_t)k * N + n]);
}

// ---------------------------------------------------------------------------
// XZ[row][1024] = Xbf16[row][K] @ WT + bias   (wave computes one 16x16 tile)
// ---------------------------------------------------------------------------
__global__ void __launch_bounds__(256)
gemm_xz_kernel(const unsigned short* __restrict__ X,
               const unsigned short* __restrict__ WT,
               const float* __restrict__ bias,
               float* __restrict__ XZ, int K, int rows)
{
    int lane = threadIdx.x & 31;
    int wv   = threadIdx.x >> 5;           // 0..7
    int rt   = blockIdx.x;
    int nt   = blockIdx.y * 8 + wv;        // 0..63
    int hi   = lane >> 4, l15 = lane & 15;
    int row  = rt * 16 + l15;
    if (row >= rows) return;
    int ncol = nt * 16 + l15;

    v8f acc;
    float bv = bias[ncol];
#pragma unroll
    for (int v = 0; v < 8; v++) acc[v] = bv;

    const unsigned short* arow = X  + (size_t)row  * K + hi * 8;
    const unsigned short* brow = WT + (size_t)ncol * K + hi * 8;
    int nk = K >> 5;
    for (int kk = 0; kk < nk; kk++) {
        FragBF a, bfb;
        a.i[0]   = *(const int4*)(arow + kk * 32);
        a.i[1]   = *(const int4*)(arow + kk * 32 + 16);
        bfb.i[0] = *(const int4*)(brow + kk * 32);
        bfb.i[1] = *(const int4*)(brow + kk * 32 + 16);
        acc = __builtin_amdgcn_wmma_f32_16x16x32_bf16(false, a.v, false, bfb.v,
                                                      (short)0, acc, false, false);
    }
    float* out = XZ + (size_t)(rt * 16 + hi * 8) * 1024 + ncol;
#pragma unroll
    for (int v = 0; v < 8; v++) out[(size_t)v * 1024] = acc[v];
}

// ---------------------------------------------------------------------------
// One workgroup per direction. 32 waves; wave (mt,jt) owns batch rows
// [16mt,16mt+16) x gate columns [16jt,16jt+16) of all four gates.
// h (bf16) lives in LDS; c (f32) lives in registers.
// ---------------------------------------------------------------------------
__global__ void __launch_bounds__(1024)
lstm_kernel(const float* __restrict__ XZf, const float* __restrict__ XZb,
            const unsigned short* __restrict__ WTf,
            const unsigned short* __restrict__ WTb,
            const unsigned char* __restrict__ mask, int tstr,
            unsigned short* __restrict__ Ybf, float* __restrict__ Yf32,
            int T, int finalLayer)
{
    __shared__ __align__(16) unsigned short hsh[32 * 256];
    __shared__ unsigned char msh[32];

    int dir = blockIdx.x;
    const float*          XZ = dir ? XZb : XZf;
    const unsigned short* WT = dir ? WTb : WTf;

    int tid  = threadIdx.x;
    int lane = tid & 31, wv = tid >> 5;
    int mt   = wv >> 4;                 // batch tile 0/1
    int jt   = wv & 15;                 // j tile 0..15
    int hi   = lane >> 4, l15 = lane & 15;
    int jcol = jt * 16 + l15;           // 0..255

    for (int i = tid; i < 32 * 256; i += 1024) hsh[i] = 0;
    float c[8];
#pragma unroll
    for (int v = 0; v < 8; v++) c[v] = 0.f;

    const unsigned short* bptr[4];
#pragma unroll
    for (int g = 0; g < 4; g++)
        bptr[g] = WT + (size_t)(g * 256 + jcol) * 256 + hi * 8;
    const unsigned short* aptr = &hsh[(mt * 16 + l15) * 256 + hi * 8];

    __syncthreads();

    for (int s = 0; s < T; s++) {
        int t = dir ? (T - 1 - s) : s;

        if (tid < 32) {                         // per-step decimated mask
            const unsigned char* mp = mask + (size_t)tid * 1696 + (size_t)t * tstr;
            unsigned char m = 1;
            for (int q = 0; q < tstr; q++) m &= (mp[q] != 0);
            msh[tid] = m;
        }

        // ---- z = xz[t] + h @ Wr  (4 gate tiles per wave) ----
        v8f acc[4];
        {
            const float* xz = XZ + ((size_t)(mt * 16 + hi * 8) * T + t) * 1024;
#pragma unroll
            for (int g = 0; g < 4; g++) {
                int n = g * 256 + jcol;
#pragma unroll
                for (int v = 0; v < 8; v++)
                    acc[g][v] = xz[(size_t)v * T * 1024 + n];
            }
#pragma unroll
            for (int kk = 0; kk < 8; kk++) {
                FragBF a;
                a.i[0] = *(const int4*)(aptr + kk * 32);
                a.i[1] = *(const int4*)(aptr + kk * 32 + 16);
#pragma unroll
                for (int g = 0; g < 4; g++) {
                    FragBF bfb;
                    bfb.i[0] = *(const int4*)(bptr[g] + kk * 32);
                    bfb.i[1] = *(const int4*)(bptr[g] + kk * 32 + 16);
                    acc[g] = __builtin_amdgcn_wmma_f32_16x16x32_bf16(
                        false, a.v, false, bfb.v, (short)0, acc[g], false, false);
                }
            }
        }
        __syncthreads();   // all h reads done, msh visible

        // ---- gates / cell update ----
#pragma unroll
        for (int v = 0; v < 8; v++) {
            int bb = mt * 16 + hi * 8 + v;
            float zi = acc[0][v], zf = acc[1][v], zg = acc[2][v], zo = acc[3][v];
            float cn = sigm(zf) * c[v] + sigm(zi) * tanhf(zg);
            float hn = sigm(zo) * tanhf(cn);
            float hv;
            if (msh[bb]) { c[v] = cn; hv = hn; }
            else         { hv = bf2f(hsh[bb * 256 + jcol]); }   // hold
            hsh[bb * 256 + jcol] = f2bf(hv);
            size_t oidx = ((size_t)bb * T + t) * 512 + (size_t)dir * 256 + jcol;
            if (finalLayer) Yf32[oidx] = hv;
            else            Ybf[oidx]  = f2bf(hv);
        }
        __syncthreads();   // h writes visible to next step
    }
}

// ---------------------------------------------------------------------------
__global__ void maskout_kernel(const unsigned char* __restrict__ mask,
                               float* __restrict__ out)
{
    int idx = blockIdx.x * blockDim.x + threadIdx.x;   // b*212 + t
    if (idx >= 32 * 212) return;
    int b = idx / 212, t = idx % 212;
    const unsigned char* mp = mask + (size_t)b * 1696 + (size_t)t * 8;
    unsigned char m = 1;
    for (int q = 0; q < 8; q++) m &= (mp[q] != 0);
    out[idx] = m ? 1.0f : 0.0f;
}

// ---------------------------------------------------------------------------
extern "C" void kernel_launch(void* const* d_in, const int* in_sizes, int n_in,
                              void* d_out, int out_size, void* d_ws, size_t ws_size,
                              hipStream_t stream)
{
    (void)in_sizes; (void)n_in; (void)out_size; (void)ws_size;

    const float*         inputs = (const float*)d_in[0];
    const unsigned char* mask   = (const unsigned char*)d_in[1];   // jnp.bool_ = 1 byte
    const float*         scale  = (const float*)d_in[2];
    const float*         dem_w  = (const float*)d_in[3];
    const float*         dem_b  = (const float*)d_in[4];

    const float* Wk[4][2] = {
        {(const float*)d_in[5],  (const float*)d_in[8]},
        {(const float*)d_in[11], (const float*)d_in[14]},
        {(const float*)d_in[17], (const float*)d_in[20]},
        {(const float*)d_in[23], (const float*)d_in[26]}};
    const float* Wr[4][2] = {
        {(const float*)d_in[6],  (const float*)d_in[9]},
        {(const float*)d_in[12], (const float*)d_in[15]},
        {(const float*)d_in[18], (const float*)d_in[21]},
        {(const float*)d_in[24], (const float*)d_in[27]}};
    const float* Bs[4][2] = {
        {(const float*)d_in[7],  (const float*)d_in[10]},
        {(const float*)d_in[13], (const float*)d_in[16]},
        {(const float*)d_in[19], (const float*)d_in[22]},
        {(const float*)d_in[25], (const float*)d_in[28]}};

    // ---- workspace carve-out ----
    char* p = (char*)d_ws;
    size_t off = 0;
    auto carve = [&](size_t bytes) -> char* {
        char* r = p + off; off += (bytes + 255) & ~(size_t)255; return r;
    };
    unsigned short* X0   = (unsigned short*)carve(54272ull * 512 * 2);
    unsigned short* Y0   = (unsigned short*)carve(54272ull * 512 * 2);
    unsigned short* Y1   = (unsigned short*)carve(27136ull * 512 * 2);
    unsigned short* Y2   = (unsigned short*)carve(13568ull * 512 * 2);
    float*          XZ0  = (float*)carve(54272ull * 1024 * 4);
    float*          XZ1  = (float*)carve(54272ull * 1024 * 4);
    unsigned short* WkT0 = (unsigned short*)carve(1024ull * 1024 * 2);
    unsigned short* WkT1 = (unsigned short*)carve(1024ull * 1024 * 2);
    unsigned short* WrT0 = (unsigned short*)carve(1024ull * 256 * 2);
    unsigned short* WrT1 = (unsigned short*)carve(1024ull * 256 * 2);

    const int Tl[4]   = {1696, 848, 424, 212};
    const int Kl[4]   = {512, 1024, 1024, 1024};
    unsigned short* Xin[4]  = {X0, Y0, Y1, Y2};
    unsigned short* Yout[4] = {Y0, Y1, Y2, nullptr};

    // ---- embed ----
    {
        int n = 32 * 1696 * 512;
        embed_kernel<<<(n + 255) / 256, 256, 0, stream>>>(
            inputs, dem_w, dem_b, scale, X0, 1696);
    }

    // ---- layers ----
    for (int l = 0; l < 4; l++) {
        int K = Kl[l], T = Tl[l], rows = 32 * T, tstr = 1 << l;

        packT_kernel<<<(K * 1024 + 255) / 256, 256, 0, stream>>>(Wk[l][0], WkT0, K, 1024);
        packT_kernel<<<(K * 1024 + 255) / 256, 256, 0, stream>>>(Wk[l][1], WkT1, K, 1024);
        packT_kernel<<<(256 * 1024 + 255) / 256, 256, 0, stream>>>(Wr[l][0], WrT0, 256, 1024);
        packT_kernel<<<(256 * 1024 + 255) / 256, 256, 0, stream>>>(Wr[l][1], WrT1, 256, 1024);

        dim3 gg(rows / 16, 8);
        gemm_xz_kernel<<<gg, 256, 0, stream>>>(Xin[l], WkT0, Bs[l][0], XZ0, K, rows);
        gemm_xz_kernel<<<gg, 256, 0, stream>>>(Xin[l], WkT1, Bs[l][1], XZ1, K, rows);

        lstm_kernel<<<2, 1024, 0, stream>>>(
            XZ0, XZ1, WrT0, WrT1, mask, tstr,
            Yout[l], (l == 3) ? (float*)d_out : nullptr, T, (l == 3) ? 1 : 0);
    }

    // ---- decimated mask tail ----
    maskout_kernel<<<(32 * 212 + 255) / 256, 256, 0, stream>>>(
        mask, (float*)d_out + (size_t)32 * 212 * 512);
}
